// PANPooling_8787503088314
// MI455X (gfx1250) — 8-bit matrix operand order repaired and checked by reading, compiled, not run on hardware
//
#include <hip/hip_runtime.h>
#include <hip/hip_bf16.h>
#include <math.h>


typedef __attribute__((ext_vector_type(8))) int v8i;

#define NN 4096
#define CC 256
#define EE 131072
#define KTOP 2048

#define WS_A    0ull
#define WS_AT   (16ull << 20)
#define WS_A2   (32ull << 20)
#define WS_A3   (48ull << 20)
#define WS_D1   (64ull << 20)
#define WS_D2   (WS_D1 + 16384ull)
#define WS_D3   (WS_D2 + 16384ull)
#define WS_SC   (WS_D3 + 16384ull)
#define WS_PERM (WS_SC + 16384ull)

#define OUT_XOUT  0
#define OUT_MPOOL 524288
#define OUT_PERM  4718592
#define OUT_SCORE 4720640

__global__ void zero_u128(uint4* __restrict__ p, size_t n16) {
    size_t i = (size_t)blockIdx.x * blockDim.x + threadIdx.x;
    size_t stride = (size_t)gridDim.x * blockDim.x;
    uint4 z; z.x = z.y = z.z = z.w = 0u;
    for (; i < n16; i += stride) p[i] = z;
}

__global__ void build_adj(const int* __restrict__ ei,
                          unsigned* __restrict__ A, unsigned* __restrict__ AT) {
    int e = blockIdx.x * blockDim.x + threadIdx.x;
    if (e >= EE) return;
    int r = ei[e];
    int c = ei[EE + e];
    size_t idx  = (size_t)r * NN + c;
    size_t idxt = (size_t)c * NN + r;
    atomicAdd(A  + (idx  >> 2), 1u << ((idx  & 3) * 8));
    atomicAdd(AT + (idxt >> 2), 1u << ((idxt & 3) * 8));
}

__global__ __launch_bounds__(256) void gemm_u8_wmma(
    const unsigned char* __restrict__ Amat,
    const unsigned char* __restrict__ BmatT,
    unsigned char* __restrict__ Out) {
    const int lane = threadIdx.x & 31;
    const int wave = threadIdx.x >> 5;
    const int row0 = blockIdx.y * 64;
    const int col0 = blockIdx.x * 512 + wave * 64;

    const int lm   = lane & 15;
    const int lk8  = (lane >> 4) * 8;
    const int lk16 = (lane >> 4) * 16;

    size_t abase[4], bbase[4];
#pragma unroll
    for (int mi = 0; mi < 4; ++mi)
        abase[mi] = (size_t)(row0 + mi * 16 + lm) * NN + lk16;
#pragma unroll
    for (int ni = 0; ni < 4; ++ni)
        bbase[ni] = (size_t)(col0 + ni * 16 + lm) * NN + lk16;

    v8i acc[4][4];
    {
        v8i z = {0, 0, 0, 0, 0, 0, 0, 0};
#pragma unroll
        for (int mi = 0; mi < 4; ++mi)
#pragma unroll
            for (int ni = 0; ni < 4; ++ni) acc[mi][ni] = z;
    }

    for (int k0 = 0; k0 < NN; k0 += 64) {
        v8i afrag[4], bfrag[4];
#pragma unroll
        for (int mi = 0; mi < 4; ++mi) {
            const unsigned char* p = Amat + abase[mi] + k0;
            uint2 q0 = *(const uint2*)(p);
            uint2 q1 = *(const uint2*)(p + 8);
            uint2 q2 = *(const uint2*)(p + 32);
            uint2 q3 = *(const uint2*)(p + 40);
            v8i a;
            a[0] = (int)q0.x; a[1] = (int)q0.y;
            a[2] = (int)q1.x; a[3] = (int)q1.y;
            a[4] = (int)q2.x; a[5] = (int)q2.y;
            a[6] = (int)q3.x; a[7] = (int)q3.y;
            afrag[mi] = a;
        }
#pragma unroll
        for (int ni = 0; ni < 4; ++ni) {
            const unsigned char* p = BmatT + bbase[ni] + k0;
            uint4 u0 = *(const uint4*)(p);
            uint4 u1 = *(const uint4*)(p + 32);
            v8i b;
            b[0] = (int)u0.x; b[1] = (int)u0.y; b[2] = (int)u0.z; b[3] = (int)u0.w;
            b[4] = (int)u1.x; b[5] = (int)u1.y; b[6] = (int)u1.z; b[7] = (int)u1.w;
            bfrag[ni] = b;
        }
#pragma unroll
        for (int mi = 0; mi < 4; ++mi)
#pragma unroll
            for (int ni = 0; ni < 4; ++ni)
                acc[mi][ni] = __builtin_amdgcn_wmma_i32_16x16x64_iu8(
                    false, afrag[mi], false, bfrag[ni], acc[mi][ni], false, false);
    }

#pragma unroll
    for (int mi = 0; mi < 4; ++mi)
#pragma unroll
        for (int ni = 0; ni < 4; ++ni)
#pragma unroll
            for (int v = 0; v < 8; ++v) {
                int r = row0 + mi * 16 + (lane >> 4) * 8 + v;
                int c = col0 + ni * 16 + lm;
                int val = acc[mi][ni][v];
                val = val < 0 ? 0 : (val > 255 ? 255 : val);
                Out[(size_t)r * NN + c] = (unsigned char)val;
            }
}

__device__ __forceinline__ int sum4b(unsigned w) {
    return (int)((w & 255u) + ((w >> 8) & 255u) + ((w >> 16) & 255u) + (w >> 24));
}

__global__ __launch_bounds__(256) void rowsum_u8(const unsigned char* __restrict__ mat,
                                                 int* __restrict__ out) {
    int row  = (blockIdx.x * 256 + threadIdx.x) >> 5;
    int lane = threadIdx.x & 31;
    const uint4* rp = (const uint4*)(mat + (size_t)row * NN);
    int s = 0;
#pragma unroll
    for (int it = 0; it < 8; ++it) {
        uint4 v = rp[it * 32 + lane];
        s += sum4b(v.x) + sum4b(v.y) + sum4b(v.z) + sum4b(v.w);
    }
    for (int off = 16; off > 0; off >>= 1) s += __shfl_down(s, off, 32);
    if (lane == 0) out[row] = s;
}

__global__ __launch_bounds__(256) void score_kernel(
    const float* __restrict__ x, const float* __restrict__ transform,
    const float* __restrict__ pfw, const float* __restrict__ ppw,
    const int* __restrict__ d1, const int* __restrict__ d2,
    const int* __restrict__ d3, float* __restrict__ score) {
    int node = (blockIdx.x * 256 + threadIdx.x) >> 5;
    int lane = threadIdx.x & 31;
    float dot = 0.f;
#pragma unroll
    for (int it = 0; it < 8; ++it) {
        int c = it * 32 + lane;
        dot += x[(size_t)node * CC + c] * transform[c];
    }
    for (int off = 16; off > 0; off >>= 1) dot += __shfl_down(dot, off, 32);
    if (lane == 0) {
        float degree = pfw[0] + pfw[1] * (float)d1[node] + pfw[2] * (float)d2[node] +
                       pfw[3] * (float)d3[node];
        score[node] = tanhf(ppw[0] * dot + ppw[1] * degree);
    }
}

__global__ void rank_kernel(const float* __restrict__ score, int* __restrict__ perm) {
    int i = blockIdx.x * blockDim.x + threadIdx.x;
    if (i >= NN) return;
    float si = score[i];
    int rank = 0;
    for (int j = 0; j < NN; ++j) {
        float sj = score[j];
        rank += (sj > si) ? 1 : ((sj == si && j < i) ? 1 : 0);
    }
    if (rank < KTOP) perm[rank] = i;
}

__global__ void xout_kernel(const float* __restrict__ x, const float* __restrict__ score,
                            const int* __restrict__ perm, float* __restrict__ out) {
    int idx = blockIdx.x * blockDim.x + threadIdx.x;
    int r = idx >> 8, c = idx & 255;
    int p = perm[r];
    out[idx] = x[(size_t)p * CC + c] * score[p];
}

__global__ void permout_kernel(const int* __restrict__ perm, const float* __restrict__ score,
                               float* __restrict__ outp, float* __restrict__ outs) {
    int r = blockIdx.x * blockDim.x + threadIdx.x;
    if (r >= KTOP) return;
    int p = perm[r];
    outp[r] = (float)p;
    outs[r] = score[p];
}

__global__ void mpool_kernel(const unsigned char* __restrict__ A,
                             const unsigned char* __restrict__ A2,
                             const unsigned char* __restrict__ A3,
                             const int* __restrict__ perm,
                             const float* __restrict__ pfw, float* __restrict__ out) {
    size_t idx = (size_t)blockIdx.x * blockDim.x + threadIdx.x;
    int a = (int)(idx >> 11), b = (int)(idx & 2047);
    int pa = perm[a], pb = perm[b];
    size_t o = (size_t)pa * NN + pb;
    float v = pfw[1] * (float)A[o] + pfw[2] * (float)A2[o] + pfw[3] * (float)A3[o];
    if (pa == pb) v += pfw[0];
    out[idx] = v;
}

extern "C" void kernel_launch(void* const* d_in, const int* in_sizes, int n_in,
                              void* d_out, int out_size, void* d_ws, size_t ws_size,
                              hipStream_t stream) {
    (void)in_sizes; (void)n_in; (void)out_size; (void)ws_size;

    const float* x         = (const float*)d_in[0];
    const int*   ei        = (const int*)d_in[1];
    const float* transform = (const float*)d_in[2];
    const float* pfw       = (const float*)d_in[3];
    const float* ppw       = (const float*)d_in[4];
    float* out = (float*)d_out;

    unsigned char* ws = (unsigned char*)d_ws;
    unsigned char* A    = ws + WS_A;
    unsigned char* AT   = ws + WS_AT;
    unsigned char* A2   = ws + WS_A2;
    unsigned char* A3   = ws + WS_A3;
    int*   d1    = (int*)(ws + WS_D1);
    int*   d2    = (int*)(ws + WS_D2);
    int*   d3    = (int*)(ws + WS_D3);
    float* score = (float*)(ws + WS_SC);
    int*   perm  = (int*)(ws + WS_PERM);

    zero_u128<<<2048, 256, 0, stream>>>((uint4*)A, (32ull << 20) / 16);
    build_adj<<<(EE + 255) / 256, 256, 0, stream>>>(ei, (unsigned*)A, (unsigned*)AT);

    dim3 gg(NN / 512, NN / 64);
    gemm_u8_wmma<<<gg, 256, 0, stream>>>(A, AT, A2);
    gemm_u8_wmma<<<gg, 256, 0, stream>>>(A2, AT, A3);

    rowsum_u8<<<512, 256, 0, stream>>>(A,  d1);
    rowsum_u8<<<512, 256, 0, stream>>>(A2, d2);
    rowsum_u8<<<512, 256, 0, stream>>>(A3, d3);

    score_kernel<<<512, 256, 0, stream>>>(x, transform, pfw, ppw, d1, d2, d3, score);
    rank_kernel<<<NN / 256, 256, 0, stream>>>(score, perm);

    xout_kernel<<<(KTOP * CC) / 256, 256, 0, stream>>>(x, score, perm, out + OUT_XOUT);
    mpool_kernel<<<(KTOP * KTOP) / 256, 256, 0, stream>>>(A, A2, A3, perm, pfw, out + OUT_MPOOL);
    permout_kernel<<<KTOP / 256, 256, 0, stream>>>(perm, score, out + OUT_PERM, out + OUT_SCORE);
}
